// BeliefPropagationTorch_45707041964165
// MI455X (gfx1250) — matrix operand order repaired and checked by reading, not compiled, not run on hardware
//
#include <hip/hip_runtime.h>
#include <hip/hip_bf16.h>

// ---------------------------------------------------------------------------
// Belief propagation on MI455X (gfx1250).
// Dominant op: inc = adj^T @ trunc(log_msg)  (8192x8192 @ 8192x2, 15 iters).
// trunc(log_msg) is exact small integers and adj is 0/1 -> f16 is lossless.
// 128MB f16 transposed adjacency fits the 192MB L2 -> iterations 2..14 run at
// L2 bandwidth. Matmul uses v_wmma_f32_16x16x32_f16 (wave32 WMMA).
// ---------------------------------------------------------------------------

#define N_NODES   2048
#define N_EDGES   8192
#define MAX_ITER  15

typedef _Float16 v16h __attribute__((ext_vector_type(16)));
typedef _Float16 v8h  __attribute__((ext_vector_type(8)));
typedef float    v8f  __attribute__((ext_vector_type(8)));

__device__ __forceinline__ float lse2(float a, float b) {
    float mx = fmaxf(a, b);
    if (mx == -INFINITY) return -INFINITY;
    return mx + logf(expf(a - mx) + expf(b - mx));
}

// ---- (1) convert msg_adj (f32, [j][i]) -> At (f16, [i][j]) via LDS transpose
__global__ __launch_bounds__(256)
void convert_transpose_f16(const float* __restrict__ adj, _Float16* __restrict__ At) {
    __shared__ _Float16 tile[32][33];
    const int tx = threadIdx.x;            // 0..31
    const int ty = threadIdx.y;            // 0..7
    const int j0 = blockIdx.x * 32;        // source row (j)
    const int i0 = blockIdx.y * 32;        // source col (i)
    #pragma unroll
    for (int r = ty; r < 32; r += 8)
        tile[tx][r] = (_Float16)adj[(size_t)(j0 + r) * N_EDGES + i0 + tx];
    __syncthreads();
    #pragma unroll
    for (int r = ty; r < 32; r += 8)
        At[(size_t)(i0 + r) * N_EDGES + j0 + tx] = tile[r][tx];
}

// ---- (2) zero log_msg and inc (iteration 0: inc = adjT @ trunc(0) = 0)
__global__ void init_zero(float* __restrict__ msg, float* __restrict__ inc) {
    int i = blockIdx.x * blockDim.x + threadIdx.x;
    if (i < 2 * N_EDGES) { msg[i] = 0.0f; inc[i] = 0.0f; }
}

// ---- (3) build B panel in WMMA lane layout: lane n=L&15 holds K=16h+0..15.
// Bpack[chunk*32 + lane][0..15] = (n<2) ? trunc16[(j0+16h+idx)*2 + n] : 0
__global__ __launch_bounds__(256)
void pack_b_panel(const _Float16* __restrict__ tru16, _Float16* __restrict__ Bpack) {
    int tid = blockIdx.x * blockDim.x + threadIdx.x;  // (E/32)*32 = 8192 threads
    if (tid >= (N_EDGES / 32) * 32) return;
    const int lane  = tid & 31;
    const int chunk = tid >> 5;
    const int n = lane & 15;
    const int h = lane >> 4;
    const int j0 = chunk * 32;
    _Float16* dst = Bpack + (size_t)tid * 16;
    #pragma unroll
    for (int idx = 0; idx < 16; ++idx) {
        int k = h * 16 + idx;
        dst[idx] = (n < 2) ? tru16[(size_t)(j0 + k) * 2 + n] : (_Float16)0.0f;
    }
}

// ---- (4) inc = At @ trunc(log_msg) with v_wmma_f32_16x16x32_f16.
// One wave per 16-row output tile; K swept over 8192 in chunks of 32.
__global__ __launch_bounds__(256)
void bp_matmul_wmma(const _Float16* __restrict__ At,
                    const _Float16* __restrict__ Bpack,
                    float* __restrict__ inc) {
    const int lane = threadIdx.x & 31;
    const int wave = threadIdx.x >> 5;
    const int tile = blockIdx.x * (blockDim.x >> 5) + wave;  // 0..511
    const int i0 = tile << 4;
    const int m = lane & 15;
    const int h = lane >> 4;

    const _Float16* arow = At + (size_t)(i0 + m) * N_EDGES;
    v8f c = {0.f, 0.f, 0.f, 0.f, 0.f, 0.f, 0.f, 0.f};

    for (int j0 = 0; j0 < N_EDGES; j0 += 32) {
        // A lane layout (16-bit A 16x32): K = 16h+0..15 (matches Bpack)
        v16h a = *(const v16h*)(arow + j0 + 16 * h);
        v16h bv = *(const v16h*)(Bpack + ((size_t)(j0 >> 5) * 32 + lane) * 16);
        // 8 args: (neg_a, A, neg_b, B, c_mod, C, reuse_a, reuse_b)
        c = __builtin_amdgcn_wmma_f32_16x16x32_f16(false, a, false, bv,
                                                   (short)0, c, false, false);
    }
    // D layout: lane L, element v -> D[8h+v][L&15]; only N<2 columns are live.
    if (m < 2) {
        const int rbase = h * 8;
        #pragma unroll
        for (int v = 0; v < 8; ++v)
            inc[(size_t)(i0 + rbase + v) * 2 + m] = c[v];
    }
}

// ---- (5) per-edge message update (damped, masked logsumexp, normalize)
__global__ __launch_bounds__(256)
void edge_update(const float* __restrict__ J, const float* __restrict__ b,
                 const int* __restrict__ msg_node, const float* __restrict__ mask,
                 const float* __restrict__ inc, float* __restrict__ msg,
                 _Float16* __restrict__ tru16) {
    int e = blockIdx.x * blockDim.x + threadIdx.x;
    if (e >= N_EDGES) return;
    const int ein  = msg_node[2 * e];
    const int eout = msg_node[2 * e + 1];
    const float Je = J[(size_t)ein * N_NODES + eout];
    const float ph = b[ein];                       // log_phi[ein] = [b, -b]
    const float t0 =  ph + inc[2 * e];             // phi_in[s2=0] + inc[s2=0]
    const float t1 = -ph + inc[2 * e + 1];         // phi_in[s2=1] + inc[s2=1]
    // log_psi[s1][s2] = Je * [[1,-1],[-1,1]]
    const float lt00 = mask[4 * e + 0] * ( Je + t0);
    const float lt01 = mask[4 * e + 1] * (-Je + t1);
    const float lt10 = mask[4 * e + 2] * (-Je + t0);
    const float lt11 = mask[4 * e + 3] * ( Je + t1);
    float n0 = 0.5f * lse2(lt00, lt01) + 0.5f * msg[2 * e];
    float n1 = 0.5f * lse2(lt10, lt11) + 0.5f * msg[2 * e + 1];
    const float z = lse2(n0, n1);
    n0 -= z; n1 -= z;
    msg[2 * e]     = n0;
    msg[2 * e + 1] = n1;
    tru16[2 * e]     = (_Float16)truncf(n0);       // exact small integers
    tru16[2 * e + 1] = (_Float16)truncf(n1);
}

// ---- (6) per-node segment-sum + log-softmax; prob_step; finals at last iter
__global__ __launch_bounds__(256)
void node_update(const float* __restrict__ b, const int* __restrict__ msg_node,
                 const float* __restrict__ msg, const float* __restrict__ target,
                 float* __restrict__ out, float* __restrict__ lossp,
                 int it, int is_final) {
    int n = blockIdx.x * blockDim.x + threadIdx.x;
    if (n >= N_NODES) return;
    float s0 = 0.0f, s1 = 0.0f;
    for (int e = 0; e < N_EDGES; ++e) {           // deterministic segment_sum
        if (msg_node[2 * e + 1] == n) { s0 += msg[2 * e]; s1 += msg[2 * e + 1]; }
    }
    float lp0 =  b[n] + s0;
    float lp1 = -b[n] + s1;
    const float z = lse2(lp0, lp1);
    lp0 -= z; lp1 -= z;
    // prob_step = prob0.T : shape (N, MAX_ITER), after probs(N*2) + loss(1)
    out[N_NODES * 2 + 1 + (size_t)n * MAX_ITER + it] = expf(lp0);
    if (is_final) {
        out[2 * n]     = expf(lp0);
        out[2 * n + 1] = expf(lp1);
        const float t0 = target[2 * n], t1 = target[2 * n + 1];
        lossp[n] = t0 * (logf(t0) - lp0) + t1 * (logf(t1) - lp1);
    }
}

// ---- (7) deterministic single-block loss reduction
__global__ __launch_bounds__(256)
void loss_reduce(const float* __restrict__ lossp, float* __restrict__ out) {
    __shared__ float sh[256];
    float s = 0.0f;
    for (int i = threadIdx.x; i < N_NODES; i += 256) s += lossp[i];
    sh[threadIdx.x] = s;
    __syncthreads();
    for (int off = 128; off > 0; off >>= 1) {
        if (threadIdx.x < off) sh[threadIdx.x] += sh[threadIdx.x + off];
        __syncthreads();
    }
    if (threadIdx.x == 0) out[N_NODES * 2] = sh[0];
}

extern "C" void kernel_launch(void* const* d_in, const int* in_sizes, int n_in,
                              void* d_out, int out_size, void* d_ws, size_t ws_size,
                              hipStream_t stream) {
    const float* J        = (const float*)d_in[0];   // N*N
    const float* b        = (const float*)d_in[1];   // N
    const int*   msg_node = (const int*)d_in[2];     // E*2
    const float* adj      = (const float*)d_in[3];   // E*E
    const float* mask     = (const float*)d_in[4];   // E*4
    const float* target   = (const float*)d_in[5];   // N*2
    float* out = (float*)d_out;                      // [N*2 probs | loss | N*15]

    // workspace carve-up (At dominates: E*E f16 = 128 MB, L2-resident)
    char* w = (char*)d_ws;
    _Float16* At    = (_Float16*)w;                                   // E*E f16
    _Float16* Bpack = (_Float16*)(w + (size_t)N_EDGES * N_EDGES * 2); // 256 KB
    float* msg   = (float*)((char*)Bpack + (size_t)N_EDGES * 16 * 2);
    float* incb  = msg + 2 * N_EDGES;
    _Float16* tru16 = (_Float16*)(incb + 2 * N_EDGES);
    float* lossp = (float*)((char*)tru16 + (size_t)2 * N_EDGES * 2);
    (void)in_sizes; (void)n_in; (void)out_size; (void)ws_size;

    convert_transpose_f16<<<dim3(N_EDGES / 32, N_EDGES / 32), dim3(32, 8), 0, stream>>>(adj, At);
    init_zero<<<(2 * N_EDGES + 255) / 256, 256, 0, stream>>>(msg, incb);

    for (int it = 0; it < MAX_ITER; ++it) {
        if (it > 0) {  // iteration 0: trunc(0)=0 -> inc stays zero, skip matmul
            pack_b_panel<<<(N_EDGES + 255) / 256, 256, 0, stream>>>(tru16, Bpack);
            bp_matmul_wmma<<<(N_EDGES / 16) / 8, 256, 0, stream>>>(At, Bpack, incb);
        }
        edge_update<<<(N_EDGES + 255) / 256, 256, 0, stream>>>(J, b, msg_node, mask,
                                                               incb, msg, tru16);
        node_update<<<(N_NODES + 255) / 256, 256, 0, stream>>>(b, msg_node, msg, target,
                                                               out, lossp, it,
                                                               it == MAX_ITER - 1);
    }
    loss_reduce<<<1, 256, 0, stream>>>(lossp, out);
}
